// SignedGraphConvolutionalNetwork_48558900249265
// MI455X (gfx1250) — matrix operand order repaired and checked by reading, not compiled, not run on hardware
//
#include <hip/hip_runtime.h>
#include <hip/hip_bf16.h>

#define NN 50000
#define DD 64
#define EE 800000
#define HH 64

typedef __attribute__((ext_vector_type(16))) __bf16 v16bf;
typedef __attribute__((ext_vector_type(8)))  float  v8f;

// ---------------------------------------------------------------------------
// Edge in-degree counts: cnt[row[e]] += 1
// ---------------------------------------------------------------------------
__global__ void count_edges_kernel(const int* __restrict__ row,
                                   float* __restrict__ cnt, int E) {
  int e = blockIdx.x * blockDim.x + threadIdx.x;
  if (e < E) atomicAdd(&cnt[row[e]], 1.0f);
}

// ---------------------------------------------------------------------------
// Scatter-add of 64-float feature rows: dst[row[e]] += src[col[e]]
// One thread handles one float4 (4 floats) of the row (16 threads per edge).
// ---------------------------------------------------------------------------
__global__ __launch_bounds__(256) void scatter_add4_kernel(
    const float* __restrict__ src, const int* __restrict__ row,
    const int* __restrict__ col, float* __restrict__ dst, int E) {
  long gid = (long)blockIdx.x * blockDim.x + threadIdx.x;
  long total = (long)E * 16;
  if (gid >= total) return;
  int e = (int)(gid >> 4);
  int q = (int)(gid & 15);
  int c = col[e];
  int r = row[e];
  float4 v = ((const float4*)src)[(long)c * 16 + q];
  float* d = dst + (long)r * 64 + q * 4;
  atomicAdd(d + 0, v.x);
  atomicAdd(d + 1, v.y);
  atomicAdd(d + 2, v.z);
  atomicAdd(d + 3, v.w);
}

// ---------------------------------------------------------------------------
// Base layer: out = tanh(l2norm([agg/cnt, x] @ W + b)), K = 128, 64 cols.
// One wave32 = one 16-row tile; 4 WMMA N-tiles of 16; K-loop of 32 (bf16).
// ---------------------------------------------------------------------------
__global__ __launch_bounds__(256) void base_layer_wmma_kernel(
    const float* __restrict__ agg, const float* __restrict__ cnt,
    const float* __restrict__ x, const float* __restrict__ W,
    const float* __restrict__ bias, float* __restrict__ out, int nrows) {
  const int lane = threadIdx.x & 31;
  const int half = lane >> 4;     // 0: lanes 0-15, 1: lanes 16-31
  const int lid  = lane & 15;
  const int tile = blockIdx.x * 8 + (threadIdx.x >> 5);
  const int row0 = tile * 16;
  if (row0 >= nrows) return;

  int rm = row0 + lid;
  if (rm > nrows - 1) rm = nrows - 1;          // clamp; store is guarded
  const float icnt = 1.0f / fmaxf(cnt[rm], 1.0f);
  const float* arow = agg + (long)rm * 64;
  const float* xrow = x   + (long)rm * 64;

  v8f acc[4];
#pragma unroll
  for (int nt = 0; nt < 4; ++nt)
#pragma unroll
    for (int v = 0; v < 8; ++v) acc[nt][v] = 0.0f;

#pragma unroll
  for (int it = 0; it < 4; ++it) {             // K = 4 * 32 = 128
    const int k0 = it * 32;
    // A fragment (16x32 bf16): lane = row; K = k0 + 8*half + e (+8 if e>=8)
    v16bf afrag;
#pragma unroll
    for (int e = 0; e < 16; ++e) {
      int k = k0 + half * 8 + e + ((e >= 8) ? 8 : 0);
      float f = (k < 64) ? arow[k] * icnt : xrow[k - 64];
      afrag[e] = (__bf16)f;
    }
#pragma unroll
    for (int nt = 0; nt < 4; ++nt) {
      // B fragment (32x16 bf16): lane = col; K = k0 + 8*half + e (+8 if e>=8)
      const int ncol = nt * 16 + lid;
      v16bf bfrag;
#pragma unroll
      for (int e = 0; e < 16; ++e) {
        int k = k0 + half * 8 + e + ((e >= 8) ? 8 : 0);
        bfrag[e] = (__bf16)W[(long)k * 64 + ncol];
      }
      acc[nt] = __builtin_amdgcn_wmma_f32_16x16x32_bf16(
          false, afrag, false, bfrag, (short)0, acc[nt], false, false);
    }
  }

  float bv[4];
#pragma unroll
  for (int nt = 0; nt < 4; ++nt) bv[nt] = bias[nt * 16 + lid];

#pragma unroll
  for (int v = 0; v < 8; ++v) {
    float tv[4];
    float ss = 0.0f;
#pragma unroll
    for (int nt = 0; nt < 4; ++nt) {
      tv[nt] = acc[nt][v] + bv[nt];
      ss += tv[nt] * tv[nt];
    }
    // reduce across the 16-lane group holding this row
    ss += __shfl_xor(ss, 1, 32);
    ss += __shfl_xor(ss, 2, 32);
    ss += __shfl_xor(ss, 4, 32);
    ss += __shfl_xor(ss, 8, 32);
    float inv = 1.0f / fmaxf(sqrtf(ss), 1e-12f);
    int r = row0 + v + half * 8;
    if (r < nrows) {
#pragma unroll
      for (int nt = 0; nt < 4; ++nt)
        out[(long)r * 64 + nt * 16 + lid] = tanhf(tv[nt] * inv);
    }
  }
}

// ---------------------------------------------------------------------------
// Deep layer: out = tanh(l2norm([aggA/cntA, aggB/cntB, x1] @ W + b)), K = 192.
// Writes into z with row stride 128 at column offset coloff.
// ---------------------------------------------------------------------------
__global__ __launch_bounds__(256) void deep_layer_wmma_kernel(
    const float* __restrict__ aggA, const float* __restrict__ cntA,
    const float* __restrict__ aggB, const float* __restrict__ cntB,
    const float* __restrict__ x1, const float* __restrict__ W,
    const float* __restrict__ bias, float* __restrict__ out,
    int coloff, int nrows) {
  const int lane = threadIdx.x & 31;
  const int half = lane >> 4;
  const int lid  = lane & 15;
  const int tile = blockIdx.x * 8 + (threadIdx.x >> 5);
  const int row0 = tile * 16;
  if (row0 >= nrows) return;

  int rm = row0 + lid;
  if (rm > nrows - 1) rm = nrows - 1;
  const float icA = 1.0f / fmaxf(cntA[rm], 1.0f);
  const float icB = 1.0f / fmaxf(cntB[rm], 1.0f);
  const float* ar = aggA + (long)rm * 64;
  const float* br = aggB + (long)rm * 64;
  const float* xr = x1   + (long)rm * 64;

  v8f acc[4];
#pragma unroll
  for (int nt = 0; nt < 4; ++nt)
#pragma unroll
    for (int v = 0; v < 8; ++v) acc[nt][v] = 0.0f;

#pragma unroll
  for (int it = 0; it < 6; ++it) {             // K = 6 * 32 = 192
    const int k0 = it * 32;
    v16bf afrag;
#pragma unroll
    for (int e = 0; e < 16; ++e) {
      int k = k0 + half * 8 + e + ((e >= 8) ? 8 : 0);
      float f;
      if (k < 64)        f = ar[k] * icA;
      else if (k < 128)  f = br[k - 64] * icB;
      else               f = xr[k - 128];
      afrag[e] = (__bf16)f;
    }
#pragma unroll
    for (int nt = 0; nt < 4; ++nt) {
      const int ncol = nt * 16 + lid;
      v16bf bfrag;
#pragma unroll
      for (int e = 0; e < 16; ++e) {
        int k = k0 + half * 8 + e + ((e >= 8) ? 8 : 0);
        bfrag[e] = (__bf16)W[(long)k * 64 + ncol];
      }
      acc[nt] = __builtin_amdgcn_wmma_f32_16x16x32_bf16(
          false, afrag, false, bfrag, (short)0, acc[nt], false, false);
    }
  }

  float bv[4];
#pragma unroll
  for (int nt = 0; nt < 4; ++nt) bv[nt] = bias[nt * 16 + lid];

#pragma unroll
  for (int v = 0; v < 8; ++v) {
    float tv[4];
    float ss = 0.0f;
#pragma unroll
    for (int nt = 0; nt < 4; ++nt) {
      tv[nt] = acc[nt][v] + bv[nt];
      ss += tv[nt] * tv[nt];
    }
    ss += __shfl_xor(ss, 1, 32);
    ss += __shfl_xor(ss, 2, 32);
    ss += __shfl_xor(ss, 4, 32);
    ss += __shfl_xor(ss, 8, 32);
    float inv = 1.0f / fmaxf(sqrtf(ss), 1e-12f);
    int r = row0 + v + half * 8;
    if (r < nrows) {
#pragma unroll
      for (int nt = 0; nt < 4; ++nt)
        out[(long)r * 128 + coloff + nt * 16 + lid] = tanhf(tv[nt] * inv);
    }
  }
}

// ---------------------------------------------------------------------------
// logits = z @ reg_W + reg_b ; log_softmax; argmax -> clarify; NLL accumulate
// ---------------------------------------------------------------------------
__global__ void logits_nll_kernel(const float* __restrict__ z,
                                  const float* __restrict__ W,
                                  const float* __restrict__ b,
                                  const int* __restrict__ comm,
                                  float* __restrict__ clar,
                                  float* __restrict__ acc, int n) {
  int i = blockIdx.x * blockDim.x + threadIdx.x;
  if (i >= n) return;
  float l0 = b[0], l1 = b[1], l2 = b[2];
  const float* zr = z + (long)i * 128;
  for (int k = 0; k < 128; ++k) {
    float zk = zr[k];
    const float* wk = W + k * 3;
    l0 += zk * wk[0];
    l1 += zk * wk[1];
    l2 += zk * wk[2];
  }
  int best = 0;
  float bm = l0;
  if (l1 > bm) { bm = l1; best = 1; }
  if (l2 > bm) { bm = l2; best = 2; }
  clar[i] = (float)best;
  float s = expf(l0 - bm) + expf(l1 - bm) + expf(l2 - bm);
  float lse = bm + logf(s);
  int c = comm[i];
  float lc = (c == 0) ? l0 : ((c == 1) ? l1 : l2);
  atomicAdd(acc, -(lc - lse));
}

// ---------------------------------------------------------------------------
// Per-edge cosine similarity losses. One wave32 per edge.
// mode 0 (pos): add max(cos,0) if clarify[i] != clarify[j]
// mode 1 (neg): add -min(cos,0) if clarify[i] == clarify[j]
// ---------------------------------------------------------------------------
__global__ __launch_bounds__(256) void edge_cos_kernel(
    const float* __restrict__ z, const float* __restrict__ clar,
    const int* __restrict__ ei, const int* __restrict__ ej,
    float* __restrict__ acc, int mode, int E) {
  int lane = threadIdx.x & 31;
  long e = (long)blockIdx.x * 8 + (threadIdx.x >> 5);
  if (e >= E) return;
  long i = ei[e], j = ej[e];
  const float* za = z + i * 128;
  const float* zb = z + j * 128;
  float dot = 0.0f, na = 0.0f, nb = 0.0f;
#pragma unroll
  for (int t = 0; t < 4; ++t) {
    float a = za[lane + 32 * t];
    float b = zb[lane + 32 * t];
    dot += a * b;
    na += a * a;
    nb += b * b;
  }
#pragma unroll
  for (int m = 1; m < 32; m <<= 1) {
    dot += __shfl_xor(dot, m, 32);
    na  += __shfl_xor(na,  m, 32);
    nb  += __shfl_xor(nb,  m, 32);
  }
  if (lane == 0) {
    float cs = dot / (fmaxf(sqrtf(na), 1e-8f) * fmaxf(sqrtf(nb), 1e-8f));
    float ci = clar[i], cj = clar[j];
    float contrib = 0.0f;
    if (mode == 0) { if (ci != cj) contrib = fmaxf(cs, 0.0f); }
    else           { if (ci == cj) contrib = -fminf(cs, 0.0f); }
    if (contrib != 0.0f) atomicAdd(acc, contrib);
  }
}

__global__ void finalize_kernel(const float* __restrict__ acc,
                                float* __restrict__ out) {
  out[0] = 0.8f * (acc[0] / (float)NN) +
           0.2f * (acc[1] / (float)EE + acc[2] / (float)EE);
}

// ---------------------------------------------------------------------------
extern "C" void kernel_launch(void* const* d_in, const int* in_sizes, int n_in,
                              void* d_out, int out_size, void* d_ws, size_t ws_size,
                              hipStream_t stream) {
  const float* X    = (const float*)d_in[0];
  const float* Wpb  = (const float*)d_in[1];   // (128,64)
  const float* bpb  = (const float*)d_in[2];
  const float* Wnb  = (const float*)d_in[3];
  const float* bnb  = (const float*)d_in[4];
  const float* Wpd  = (const float*)d_in[5];   // (192,64)
  const float* bpd  = (const float*)d_in[6];
  const float* Wnd  = (const float*)d_in[7];
  const float* bnd  = (const float*)d_in[8];
  const float* regW = (const float*)d_in[9];   // (128,3)
  const float* regb = (const float*)d_in[10];
  const int*   posE = (const int*)d_in[11];    // [2,E]
  const int*   negE = (const int*)d_in[12];
  const int*   comm = (const int*)d_in[13];

  const int* rp = posE;        const int* cp = posE + EE;
  const int* rn = negE;        const int* cn = negE + EE;

  // workspace layout (floats)
  float* acc  = (float*)d_ws;                       // [16] loss accumulators
  float* cntP = acc  + 16;                          // [N]
  float* cntN = cntP + NN;                          // [N]
  float* aggA = cntN + NN;                          // [N*64]
  float* aggB = aggA + (size_t)NN * 64;             // [N*64]
  float* hP   = aggB + (size_t)NN * 64;             // [N*64]
  float* hN   = hP   + (size_t)NN * 64;             // [N*64]

  float* out  = (float*)d_out;
  float* z    = out + 1;                            // [N*128]
  float* clar = z + (size_t)NN * 128;               // [N]

  const size_t featBytes = (size_t)NN * 64 * sizeof(float);
  dim3 blk(256);
  const int eBlocks  = (EE + 255) / 256;
  const int sBlocks  = (int)(((long)EE * 16 + 255) / 256);
  const int gBlocks  = (NN + 127) / 128;            // 8 row-tiles per block
  const int nBlocks  = (NN + 255) / 256;
  const int wBlocks  = (EE + 7) / 8;                // one wave per edge

  // zero accumulators / counts / agg buffers
  hipMemsetAsync(acc,  0, 16 * sizeof(float), stream);
  hipMemsetAsync(cntP, 0, NN * sizeof(float), stream);
  hipMemsetAsync(cntN, 0, NN * sizeof(float), stream);
  hipMemsetAsync(aggA, 0, featBytes, stream);
  hipMemsetAsync(aggB, 0, featBytes, stream);

  // in-degree counts (shared by both layers)
  count_edges_kernel<<<eBlocks, blk, 0, stream>>>(rp, cntP, EE);
  count_edges_kernel<<<eBlocks, blk, 0, stream>>>(rn, cntN, EE);

  // layer 0: aggregate X over pos / neg edges, then WMMA base layers
  scatter_add4_kernel<<<sBlocks, blk, 0, stream>>>(X, rp, cp, aggA, EE);
  scatter_add4_kernel<<<sBlocks, blk, 0, stream>>>(X, rn, cn, aggB, EE);
  base_layer_wmma_kernel<<<gBlocks, blk, 0, stream>>>(aggA, cntP, X, Wpb, bpb, hP, NN);
  base_layer_wmma_kernel<<<gBlocks, blk, 0, stream>>>(aggB, cntN, X, Wnb, bnb, hN, NN);

  // layer 1, h_pos2: meanP(h_pos), meanN(h_neg), x1 = h_pos
  hipMemsetAsync(aggA, 0, featBytes, stream);
  hipMemsetAsync(aggB, 0, featBytes, stream);
  scatter_add4_kernel<<<sBlocks, blk, 0, stream>>>(hP, rp, cp, aggA, EE);
  scatter_add4_kernel<<<sBlocks, blk, 0, stream>>>(hN, rn, cn, aggB, EE);
  deep_layer_wmma_kernel<<<gBlocks, blk, 0, stream>>>(aggA, cntP, aggB, cntN,
                                                      hP, Wpd, bpd, z, 0, NN);

  // layer 1, h_neg2: meanP(h_neg), meanN(h_pos), x1 = h_neg
  hipMemsetAsync(aggA, 0, featBytes, stream);
  hipMemsetAsync(aggB, 0, featBytes, stream);
  scatter_add4_kernel<<<sBlocks, blk, 0, stream>>>(hN, rp, cp, aggA, EE);
  scatter_add4_kernel<<<sBlocks, blk, 0, stream>>>(hP, rn, cn, aggB, EE);
  deep_layer_wmma_kernel<<<gBlocks, blk, 0, stream>>>(aggA, cntP, aggB, cntN,
                                                      hN, Wnd, bnd, z, 64, NN);

  // classification head + losses
  logits_nll_kernel<<<nBlocks, blk, 0, stream>>>(z, regW, regb, comm, clar, acc + 0, NN);
  edge_cos_kernel<<<wBlocks, blk, 0, stream>>>(z, clar, rp, cp, acc + 1, 0, EE);
  edge_cos_kernel<<<wBlocks, blk, 0, stream>>>(z, clar, rn, cn, acc + 2, 1, EE);
  finalize_kernel<<<1, 1, 0, stream>>>(acc, out);
}
